// attn_36867999269116
// MI455X (gfx1250) — matrix operand order repaired and checked by reading, not compiled, not run on hardware
//
#include <hip/hip_runtime.h>
#include <hip/hip_bf16.h>

#define BS   2
#define SL   2048
#define DM   2048
#define DQ   1536
#define DKV  512
#define NH   16
#define HD   128
#define NT   (BS * SL)

typedef __bf16 bf16_t;
typedef __bf16 v16bf __attribute__((ext_vector_type(16)));
typedef float  v8f   __attribute__((ext_vector_type(8)));

static __device__ __forceinline__ v8f wmma_bf16(v16bf a, v16bf b, v8f c) {
  // (neg_a, A, neg_b, B, c_mod, C, reuse_a, reuse_b)
  return __builtin_amdgcn_wmma_f32_16x16x32_bf16(false, a, false, b, (short)0, c, false, false);
}

// A-matrix (16x32 bf16) per-lane K base for VGPR v, lane-half h:
// lanes 0-15 hold K {0..7,16..23}, lanes 16-31 hold K {8..15,24..31}
static __device__ __forceinline__ int a_kbase(int v, int half) {
  return (v < 4 ? 2 * v : 16 + 2 * (v - 4)) + 8 * half;
}

union FragU { v16bf v; unsigned u[8]; bf16_t h[16]; };

// LDS byte offset of a __shared__ object: generic LDS address low 32 bits
static __device__ __forceinline__ unsigned lds_addr(const void* p) {
  return (unsigned)(unsigned long long)p;
}

// Async global -> LDS 16B copy (GVS mode: saddr base + 32-bit vgpr byte offset).
// Tracked by ASYNCcnt; completion enforced with s_wait_asynccnt.
static __device__ __forceinline__ void async_b128(unsigned lds_off, unsigned goff,
                                                  const void* base) {
  asm volatile("global_load_async_to_lds_b128 %0, %1, %2"
               :: "v"(lds_off), "v"(goff), "s"(base) : "memory");
}
static __device__ __forceinline__ void wait_async0() {
  asm volatile("s_wait_asynccnt 0x0" ::: "memory");
}

// ---------------- fp32 -> bf16 cast ----------------
__global__ void k_cast_bf16(const float* __restrict__ s, bf16_t* __restrict__ d, int n4) {
  int i = blockIdx.x * blockDim.x + threadIdx.x;
  if (i >= n4) return;
  float4 f = ((const float4*)s)[i];
  alignas(8) bf16_t o[4] = { (bf16_t)f.x, (bf16_t)f.y, (bf16_t)f.z, (bf16_t)f.w };
  ((uint2*)d)[i] = *(const uint2*)o;
}

// ---------------- RoPE + residual add -> bf16 ----------------
// tn[j] = dim / theta^(2j) per the reference; dst = ct + rope(rr)
__global__ void k_rope_add(const float* __restrict__ ct, const float* __restrict__ rr,
                           bf16_t* __restrict__ dst, int npairs) {
  int i = blockIdx.x * blockDim.x + threadIdx.x;
  if (i >= npairs) return;
  int j = i & 1023;        // pair index within dim (2048 -> 1024 pairs)
  int t = i >> 10;         // flat token
  int s = t & (SL - 1);    // sequence position
  float tn  = 2048.0f * __expf(-2.0f * (float)j * 9.210340371976184f); // ln(10000)
  float ang = (float)s * tn;
  float sn, cs;
  __sincosf(ang, &sn, &cs);
  float xe = rr[2 * i], xo = rr[2 * i + 1];
  dst[2 * i]     = (bf16_t)(ct[2 * i]     + xe * cs - xo * sn);
  dst[2 * i + 1] = (bf16_t)(ct[2 * i + 1] + xe * sn + xo * cs);
}

// ---------------- tiled WMMA GEMM: C[M,N] = A[M,K] * W[N,K]^T + bias ----------------
// 128x128 block tile, K staged 64 deep, double-buffered via async global->LDS.
#define KTILE 64
#define ASTR  72   // bf16 row stride (144B rows: 16B aligned, bank-skewed)

template <bool OUT_BF16>
__global__ __launch_bounds__(256)
void k_gemm(const bf16_t* __restrict__ A, const bf16_t* __restrict__ W,
            const float* __restrict__ bias, void* __restrict__ Cout,
            int M, int N, int K) {
  __shared__ bf16_t As[2][128 * ASTR];
  __shared__ bf16_t Bs[2][128 * ASTR];

  const int tid  = threadIdx.x;
  const int lane = tid & 31;
  const int wave = tid >> 5;
  const int half = lane >> 4;
  const int l16  = lane & 15;
  const int wm   = wave & 1;   // 2 wave-rows (64 M each)
  const int wn   = wave >> 1;  // 4 wave-cols (32 N each)
  const int mblk = blockIdx.y * 128;
  const int nblk = blockIdx.x * 128;

  const unsigned asb[2] = { lds_addr(&As[0][0]), lds_addr(&As[1][0]) };
  const unsigned bsb[2] = { lds_addr(&Bs[0][0]), lds_addr(&Bs[1][0]) };

  // stage one 128xKTILE tile of A and W into LDS buffer `buf` asynchronously
  auto stage = [&](int kb, int buf) {
#pragma unroll
    for (int i = 0; i < 4; ++i) {
      int c   = tid * 4 + i;          // 0..1023 chunks of 8 bf16
      int row = c >> 3;
      int ko  = (c & 7) * 8;
      unsigned loff = (unsigned)((row * ASTR + ko) * 2);
      unsigned gA   = ((unsigned)(mblk + row) * (unsigned)K + (unsigned)(kb + ko)) * 2u;
      unsigned gB   = ((unsigned)(nblk + row) * (unsigned)K + (unsigned)(kb + ko)) * 2u;
      async_b128(asb[buf] + loff, gA, A);
      async_b128(bsb[buf] + loff, gB, W);
    }
  };

  v8f acc[4][2];
#pragma unroll
  for (int i = 0; i < 4; ++i)
#pragma unroll
    for (int j = 0; j < 2; ++j) acc[i][j] = 0.0f;

  stage(0, 0);
  int cur = 0;
  for (int kb = 0; kb < K; kb += KTILE) {
    wait_async0();        // own async writes (this buffer's share) complete
    __syncthreads();      // everyone's writes complete; prev reads complete
    if (kb + KTILE < K) stage(kb + KTILE, cur ^ 1);  // overlap with compute

    const bf16_t* Asc = As[cur];
    const bf16_t* Bsc = Bs[cur];
#pragma unroll
    for (int ks = 0; ks < KTILE; ks += 32) {
      v16bf af[4], bfm[2];
#pragma unroll
      for (int i = 0; i < 4; ++i) {
        int row = wm * 64 + i * 16 + l16;
        FragU f;
#pragma unroll
        for (int v = 0; v < 8; ++v)
          f.u[v] = *(const unsigned*)&Asc[row * ASTR + ks + a_kbase(v, half)];
        af[i] = f.v;
      }
#pragma unroll
      for (int j = 0; j < 2; ++j) {
        int col = wn * 32 + j * 16 + l16;
        FragU f;
#pragma unroll
        for (int v = 0; v < 8; ++v)  // B: lane=col, K = a_kbase(v, half)(+1)
          f.u[v] = *(const unsigned*)&Bsc[col * ASTR + ks + a_kbase(v, half)];
        bfm[j] = f.v;
      }
#pragma unroll
      for (int i = 0; i < 4; ++i)
#pragma unroll
        for (int j = 0; j < 2; ++j)
          acc[i][j] = wmma_bf16(af[i], bfm[j], acc[i][j]);
    }
    cur ^= 1;
  }

  // epilogue: C layout VGPR r -> row r (lanes 0-15) / r+8 (lanes 16-31), col = lane&15
#pragma unroll
  for (int i = 0; i < 4; ++i)
#pragma unroll
    for (int j = 0; j < 2; ++j) {
      int colg = nblk + wn * 32 + j * 16 + l16;
      float bv = bias[colg];
#pragma unroll
      for (int r = 0; r < 8; ++r) {
        int rowg = mblk + wm * 64 + i * 16 + r + 8 * half;
        float vv = acc[i][j][r] + bv;
        if (OUT_BF16) ((bf16_t*)Cout)[(size_t)rowg * N + colg] = (bf16_t)vv;
        else          ((float*)Cout)[(size_t)rowg * N + colg]  = vv;
      }
    }
}

// ---------------- flash attention, 16 heads x 128 dim ----------------
#define KSTR 136  // 128 + 8 pad (272B rows, 16B aligned)
#define PSTR 34

__global__ __launch_bounds__(256)
void k_attn(const bf16_t* __restrict__ Q, const bf16_t* __restrict__ Kb,
            const bf16_t* __restrict__ Vb, bf16_t* __restrict__ O) {
  __shared__ bf16_t Ks[2][32 * KSTR];
  __shared__ bf16_t Vs[2][32 * KSTR];
  __shared__ bf16_t Ps[8 * 16 * PSTR];  // per-wave P transpose staging

  const int tid  = threadIdx.x;
  const int lane = tid & 31;
  const int wave = tid >> 5;
  const int half = lane >> 4;
  const int l16  = lane & 15;
  const int bh   = blockIdx.y;       // b*NH + h
  const int b    = bh >> 4;
  const int h    = bh & 15;
  const int q0   = blockIdx.x * 128 + wave * 16;

  const bf16_t* qg = Q  + (size_t)b * SL * DM + (size_t)h * HD;
  const bf16_t* kg = Kb + (size_t)b * SL * DM + (size_t)h * HD;
  const bf16_t* vg = Vb + (size_t)b * SL * DM + (size_t)h * HD;

  const unsigned ksb[2] = { lds_addr(&Ks[0][0]), lds_addr(&Ks[1][0]) };
  const unsigned vsb[2] = { lds_addr(&Vs[0][0]), lds_addr(&Vs[1][0]) };

  // async stage of one 32x128 K tile and V tile
  auto stage = [&](int kb, int buf) {
#pragma unroll
    for (int i = 0; i < 2; ++i) {
      int c   = tid * 2 + i;          // 0..511 chunks of 8 bf16
      int row = c >> 4;
      int ko  = (c & 15) * 8;
      unsigned loff = (unsigned)((row * KSTR + ko) * 2);
      unsigned goff = ((unsigned)(kb + row) * (unsigned)DM + (unsigned)ko) * 2u;
      async_b128(ksb[buf] + loff, goff, kg);
      async_b128(vsb[buf] + loff, goff, vg);
    }
  };

  // Q fragments: 4 x (16x32) A-tiles covering head dim 128, loaded once
  v16bf qf[4];
#pragma unroll
  for (int kc = 0; kc < 4; ++kc) {
    FragU f;
    int row = q0 + l16;
#pragma unroll
    for (int v = 0; v < 8; ++v)
      f.u[v] = *(const unsigned*)&qg[(size_t)row * DM + kc * 32 + a_kbase(v, half)];
    qf[kc] = f.v;
  }

  v8f acc[8];
#pragma unroll
  for (int n = 0; n < 8; ++n) acc[n] = 0.0f;
  float m_[8], l_[8];
#pragma unroll
  for (int r = 0; r < 8; ++r) { m_[r] = -1e30f; l_[r] = 0.0f; }
  const float scale = 0.08838834764831845f;  // 1/sqrt(128)

  stage(0, 0);
  int cur = 0;
  for (int kb = 0; kb < SL; kb += 32) {
    wait_async0();
    __syncthreads();
    if (kb + 32 < SL) stage(kb + 32, cur ^ 1);  // overlap next tile with compute

    const bf16_t* Ksc = Ks[cur];
    const bf16_t* Vsc = Vs[cur];

    // S = Q * K^T  (two 16x16 key tiles)
    v8f s0 = 0.0f, s1 = 0.0f;
#pragma unroll
    for (int kc = 0; kc < 4; ++kc) {
      FragU f0, f1;
#pragma unroll
      for (int v = 0; v < 8; ++v) {
        int koff = kc * 32 + a_kbase(v, half);           // B: K along VGPRs
        f0.u[v] = *(const unsigned*)&Ksc[(l16)      * KSTR + koff];  // keys 0..15
        f1.u[v] = *(const unsigned*)&Ksc[(16 + l16) * KSTR + koff];  // keys 16..31
      }
      s0 = wmma_bf16(qf[kc], f0.v, s0);
      s1 = wmma_bf16(qf[kc], f1.v, s1);
    }

    // online softmax: rows live in VGPR index r across each 16-lane half-group
    float alpha[8];
#pragma unroll
    for (int r = 0; r < 8; ++r) {
      float a0 = s0[r] * scale, a1 = s1[r] * scale;
      float vmax = fmaxf(a0, a1);
#pragma unroll
      for (int off = 1; off < 16; off <<= 1)
        vmax = fmaxf(vmax, __shfl_xor(vmax, off, 32));
      float mn = fmaxf(m_[r], vmax);
      float al = __expf(m_[r] - mn);
      float p0 = __expf(a0 - mn), p1 = __expf(a1 - mn);
      float rs = p0 + p1;
#pragma unroll
      for (int off = 1; off < 16; off <<= 1)
        rs += __shfl_xor(rs, off, 32);
      l_[r] = l_[r] * al + rs;
      m_[r] = mn;
      alpha[r] = al;
      // scatter P into per-wave LDS pad (C-layout -> row-major); wave-local,
      // ordered by hardware DScnt (no block barrier needed)
      bf16_t* pr = &Ps[(wave * 16 + r + 8 * half) * PSTR];
      pr[l16]      = (bf16_t)p0;
      pr[16 + l16] = (bf16_t)p1;
    }
#pragma unroll
    for (int n = 0; n < 8; ++n)
#pragma unroll
      for (int r = 0; r < 8; ++r) acc[n][r] *= alpha[r];

    // reload P as one 16x32 A fragment (same wave wrote it)
    FragU pf;
#pragma unroll
    for (int v = 0; v < 8; ++v)
      pf.u[v] = *(const unsigned*)&Ps[(wave * 16 + l16) * PSTR + a_kbase(v, half)];

    // O += P * V  (8 dim tiles of 16)
#pragma unroll
    for (int n = 0; n < 8; ++n) {
      FragU fv;
#pragma unroll
      for (int v = 0; v < 8; ++v) {
        int key = a_kbase(v, half);
        fv.h[2 * v]     = Vsc[(key)     * KSTR + n * 16 + l16];
        fv.h[2 * v + 1] = Vsc[(key + 1) * KSTR + n * 16 + l16];
      }
      acc[n] = wmma_bf16(pf.v, fv.v, acc[n]);
    }
    cur ^= 1;
  }

  // normalize and write bf16 output rows
#pragma unroll
  for (int n = 0; n < 8; ++n)
#pragma unroll
    for (int r = 0; r < 8; ++r) {
      int rowg = q0 + r + 8 * half;
      float ov = acc[n][r] / l_[r];
      O[((size_t)b * SL + rowg) * DM + h * HD + n * 16 + l16] = (bf16_t)ov;
    }
}

// ---------------- host orchestration ----------------
extern "C" void kernel_launch(void* const* d_in, const int* in_sizes, int n_in,
                              void* d_out, int out_size, void* d_ws, size_t ws_size,
                              hipStream_t stream) {
  (void)in_sizes; (void)n_in; (void)out_size; (void)ws_size;
  const float* x     = (const float*)d_in[0];
  const float* w_dq  = (const float*)d_in[1];
  const float* b_dq  = (const float*)d_in[2];
  const float* w_uq  = (const float*)d_in[3];
  const float* b_uq  = (const float*)d_in[4];
  const float* w_dkv = (const float*)d_in[5];
  const float* b_dkv = (const float*)d_in[6];
  const float* w_uk  = (const float*)d_in[7];
  const float* b_uk  = (const float*)d_in[8];
  const float* w_uv  = (const float*)d_in[9];
  const float* b_uv  = (const float*)d_in[10];
  const float* w_qr  = (const float*)d_in[11];
  const float* b_qr  = (const float*)d_in[12];
  const float* w_kr  = (const float*)d_in[13];
  const float* b_kr  = (const float*)d_in[14];
  const float* w_o   = (const float*)d_in[15];
  const float* b_o   = (const float*)d_in[16];

  char* p = (char*)d_ws;
  auto take = [&](size_t bytes) { void* r = (void*)p; p += (bytes + 255) & ~(size_t)255; return r; };
  bf16_t* xb   = (bf16_t*)take((size_t)NT  * DM  * 2);
  bf16_t* wdq  = (bf16_t*)take((size_t)DQ  * DM  * 2);
  bf16_t* wuq  = (bf16_t*)take((size_t)DM  * DQ  * 2);
  bf16_t* wdkv = (bf16_t*)take((size_t)DKV * DM  * 2);
  bf16_t* wuk  = (bf16_t*)take((size_t)DM  * DKV * 2);
  bf16_t* wuv  = (bf16_t*)take((size_t)DM  * DKV * 2);
  bf16_t* wqr  = (bf16_t*)take((size_t)DM  * DQ  * 2);
  bf16_t* wkr  = (bf16_t*)take((size_t)DM  * DKV * 2);
  bf16_t* wob  = (bf16_t*)take((size_t)DM  * DM  * 2);
  bf16_t* ctq  = (bf16_t*)take((size_t)NT  * DQ  * 2);
  bf16_t* ctkv = (bf16_t*)take((size_t)NT  * DKV * 2);
  float*  qct  = (float*) take((size_t)NT  * DM  * 4);
  float*  qr   = (float*) take((size_t)NT  * DM  * 4);
  float*  kct  = (float*) take((size_t)NT  * DM  * 4);
  float*  kr   = (float*) take((size_t)NT  * DM  * 4);
  bf16_t* qb   = (bf16_t*)take((size_t)NT  * DM  * 2);
  bf16_t* kb   = (bf16_t*)take((size_t)NT  * DM  * 2);
  bf16_t* vb   = (bf16_t*)take((size_t)NT  * DM  * 2);
  bf16_t* ab   = (bf16_t*)take((size_t)NT  * DM  * 2);

  auto cast = [&](const float* src, bf16_t* dst, size_t n) {
    int n4 = (int)(n / 4);
    k_cast_bf16<<<dim3((n4 + 255) / 256), dim3(256), 0, stream>>>(src, dst, n4);
  };
  cast(x,     xb,   (size_t)NT  * DM);
  cast(w_dq,  wdq,  (size_t)DQ  * DM);
  cast(w_uq,  wuq,  (size_t)DM  * DQ);
  cast(w_dkv, wdkv, (size_t)DKV * DM);
  cast(w_uk,  wuk,  (size_t)DM  * DKV);
  cast(w_uv,  wuv,  (size_t)DM  * DKV);
  cast(w_qr,  wqr,  (size_t)DM  * DQ);
  cast(w_kr,  wkr,  (size_t)DM  * DKV);
  cast(w_o,   wob,  (size_t)DM  * DM);

  auto gemm = [&](const bf16_t* A, const bf16_t* W, const float* bias, void* C,
                  int M, int N, int K, bool outbf) {
    dim3 g(N / 128, M / 128);
    if (outbf) k_gemm<true ><<<g, dim3(256), 0, stream>>>(A, W, bias, C, M, N, K);
    else       k_gemm<false><<<g, dim3(256), 0, stream>>>(A, W, bias, C, M, N, K);
  };

  gemm(xb,   wdq,  b_dq,  ctq,  NT, DQ,  DM,  true);   // ct_q   (bf16)
  gemm(xb,   wdkv, b_dkv, ctkv, NT, DKV, DM,  true);   // ct_kv  (bf16)
  gemm(ctq,  wuq,  b_uq,  qct,  NT, DM,  DQ,  false);  // q_ct   (f32)
  gemm(ctq,  wqr,  b_qr,  qr,   NT, DM,  DQ,  false);  // rope_q raw (f32)
  gemm(ctkv, wuk,  b_uk,  kct,  NT, DM,  DKV, false);  // k_ct   (f32)
  gemm(ctkv, wkr,  b_kr,  kr,   NT, DM,  DKV, false);  // rope_k raw (f32)
  gemm(ctkv, wuv,  b_uv,  vb,   NT, DM,  DKV, true);   // v      (bf16)

  int npairs = NT * (DM / 2);
  k_rope_add<<<dim3((npairs + 255) / 256), dim3(256), 0, stream>>>(qct, qr, qb, npairs);
  k_rope_add<<<dim3((npairs + 255) / 256), dim3(256), 0, stream>>>(kct, kr, kb, npairs);

  k_attn<<<dim3(SL / 128, BS * NH), dim3(256), 0, stream>>>(qb, kb, vb, ab);

  gemm(ab, wob, b_o, d_out, NT, DM, DM, false);        // final projection (f32)
}
